// VisionTransformerEncoderCell_39926015983744
// MI455X (gfx1250) — matrix operand order repaired and checked by reading, not compiled, not run on hardware
//
#include <hip/hip_runtime.h>
#include <hip/hip_bf16.h>

typedef _Float16 h16;
typedef __attribute__((ext_vector_type(16))) _Float16 v16h;
typedef __attribute__((ext_vector_type(8)))  _Float16 v8h;
typedef __attribute__((ext_vector_type(8)))  float    v8f;

#define BB 8
#define NN 1024
#define CC_ 768
#define HH 12
#define DH 64
#define FF_ 3072

__device__ __forceinline__ v16h mk16(v8h a, v8h b) {
    v16h r;
#pragma unroll
    for (int i = 0; i < 8; ++i) { r[i] = a[i]; r[i + 8] = b[i]; }
    return r;
}

__device__ __forceinline__ v8f wmma16(v16h a, v16h b, v8f c) {
    return __builtin_amdgcn_wmma_f32_16x16x32_f16(false, a, false, b, (short)0, c, false, false);
}

// ds_swizzle XOR butterfly within 16-lane groups (and_mask=0x1f, xor in bits[14:10])
template <int IMM>
__device__ __forceinline__ float swz(float v) {
    return __int_as_float(__builtin_amdgcn_ds_swizzle(__float_as_int(v), IMM));
}
__device__ __forceinline__ float grp_max16(float v) {
    v = fmaxf(v, swz<0x041F>(v));   // xor 1
    v = fmaxf(v, swz<0x081F>(v));   // xor 2
    v = fmaxf(v, swz<0x101F>(v));   // xor 4
    v = fmaxf(v, swz<0x201F>(v));   // xor 8
    return v;
}
__device__ __forceinline__ float grp_sum16(float v) {
    v += swz<0x041F>(v);
    v += swz<0x081F>(v);
    v += swz<0x101F>(v);
    v += swz<0x201F>(v);
    return v;
}

// ---------------- weight transpose-cast: in[K][N] f32 -> out[N][K] f16 ----------------
__global__ __launch_bounds__(256) void castT_kernel(const float* __restrict__ in,
                                                    h16* __restrict__ out,
                                                    int K, int N) {
    int idx = blockIdx.x * 256 + threadIdx.x;
    if (idx >= K * N) return;
    int k = idx / N;
    int n = idx - k * N;
    out[(size_t)n * K + k] = (h16)in[idx];
}

// ---------------- layernorm over C=768: f32 in -> f16 out ----------------
__global__ __launch_bounds__(256) void ln_kernel(const float* __restrict__ x,
                                                 const float* __restrict__ g,
                                                 const float* __restrict__ b,
                                                 h16* __restrict__ out) {
    __shared__ float s1[256], s2[256];
    const int row = blockIdx.x;
    const int tid = threadIdx.x;
    const float* xr = x + (size_t)row * CC_;
    float v0 = xr[tid], v1 = xr[tid + 256], v2 = xr[tid + 512];
    s1[tid] = v0 + v1 + v2;
    s2[tid] = v0 * v0 + v1 * v1 + v2 * v2;
    __syncthreads();
    for (int off = 128; off > 0; off >>= 1) {
        if (tid < off) { s1[tid] += s1[tid + off]; s2[tid] += s2[tid + off]; }
        __syncthreads();
    }
    const float mu = s1[0] * (1.0f / CC_);
    const float var = s2[0] * (1.0f / CC_) - mu * mu;
    const float rinv = rsqrtf(var + 1e-5f);
    h16* orow = out + (size_t)row * CC_;
    orow[tid]       = (h16)((v0 - mu) * rinv * g[tid]       + b[tid]);
    orow[tid + 256] = (h16)((v1 - mu) * rinv * g[tid + 256] + b[tid + 256]);
    orow[tid + 512] = (h16)((v2 - mu) * rinv * g[tid + 512] + b[tid + 512]);
}

// ---------------- GEMM: Y[R,N] = A[R,K](f16) * Wt[N,K](f16)^T + bias ----------------
// per-wave tile: 32 rows x 64 cols (2 A-frags x 4 B-frags -> 8 WMMA / K-step)
#define MODE_F16       0
#define MODE_F16_RELU  1
#define MODE_TRANS_F16 2   // write out[(b*768+col)*1024 + token]  (for V)
#define MODE_F32_RES   3   // write f32 out = acc + bias + res

template <int MODE>
__global__ __launch_bounds__(128) void gemm_kernel(const h16* __restrict__ A,
                                                   const h16* __restrict__ Wt,
                                                   const float* __restrict__ bias,
                                                   h16* __restrict__ outh,
                                                   float* __restrict__ outf,
                                                   const float* __restrict__ res,
                                                   int K, int N) {
    const int lane = threadIdx.x & 31;
    const int wv   = threadIdx.x >> 5;
    const int m    = lane & 15;
    const int kg   = lane >> 4;
    const int r0   = blockIdx.x * 32;
    const int n0   = (blockIdx.y * 4 + wv) * 64;

    const h16* arow0 = A + (size_t)(r0 + m) * K + 16 * kg;
    const h16* arow1 = A + (size_t)(r0 + 16 + m) * K + 16 * kg;
    const h16* bp0 = Wt + (size_t)(n0 +  0 + m) * K + 16 * kg;
    const h16* bp1 = Wt + (size_t)(n0 + 16 + m) * K + 16 * kg;
    const h16* bp2 = Wt + (size_t)(n0 + 32 + m) * K + 16 * kg;
    const h16* bp3 = Wt + (size_t)(n0 + 48 + m) * K + 16 * kg;

    v8f a00 = {}, a01 = {}, a02 = {}, a03 = {};
    v8f a10 = {}, a11 = {}, a12 = {}, a13 = {};
    for (int kb = 0; kb < K; kb += 32) {
        v16h af0 = mk16(*(const v8h*)(arow0 + kb), *(const v8h*)(arow0 + kb + 8));
        v16h af1 = mk16(*(const v8h*)(arow1 + kb), *(const v8h*)(arow1 + kb + 8));
        v16h bf0 = *(const v16h*)(bp0 + kb);
        v16h bf1 = *(const v16h*)(bp1 + kb);
        v16h bf2 = *(const v16h*)(bp2 + kb);
        v16h bf3 = *(const v16h*)(bp3 + kb);
        a00 = wmma16(af0, bf0, a00);
        a10 = wmma16(af1, bf0, a10);
        a01 = wmma16(af0, bf1, a01);
        a11 = wmma16(af1, bf1, a11);
        a02 = wmma16(af0, bf2, a02);
        a12 = wmma16(af1, bf2, a12);
        a03 = wmma16(af0, bf3, a03);
        a13 = wmma16(af1, bf3, a13);
    }

    v8f accs[2][4] = {{a00, a01, a02, a03}, {a10, a11, a12, a13}};
#pragma unroll
    for (int gR = 0; gR < 2; ++gR) {
#pragma unroll
        for (int s = 0; s < 4; ++s) {
            const int col = n0 + s * 16 + m;
            const float bv = bias[col];
#pragma unroll
            for (int j = 0; j < 8; ++j) {
                float val = accs[gR][s][j] + bv;
                const int row = r0 + 16 * gR + j + 8 * kg;
                if (MODE == MODE_F16) {
                    outh[(size_t)row * N + col] = (h16)val;
                } else if (MODE == MODE_F16_RELU) {
                    outh[(size_t)row * N + col] = (h16)fmaxf(val, 0.0f);
                } else if (MODE == MODE_TRANS_F16) {
                    const int bat = row >> 10;        // token rows per batch = 1024
                    const int tok = row & 1023;
                    outh[((size_t)(bat * CC_ + col)) * NN + tok] = (h16)val;
                } else { // MODE_F32_RES
                    outf[(size_t)row * N + col] = val + res[(size_t)row * N + col];
                }
            }
        }
    }
}

// ---------------- flash attention: one wave per (b, h, 16-query tile) ----------------
__global__ __launch_bounds__(128) void attn_kernel(const h16* __restrict__ q,
                                                   const h16* __restrict__ k,
                                                   const h16* __restrict__ vT,
                                                   h16* __restrict__ z) {
    __shared__ __align__(16) _Float16 plds[4 * 512];
    const int wv   = threadIdx.x >> 5;
    const int lane = threadIdx.x & 31;
    const int m    = lane & 15;
    const int kg   = lane >> 4;
    h16* myp = plds + wv * 512;   // 16x32 f16 tile, row-major

    const int gid = blockIdx.x * 4 + wv;          // 0..6143
    const int qt  = gid & 63;
    const int bh  = gid >> 6;
    const int h   = bh % HH;
    const int b   = bh / HH;
    const int q0  = qt * 16;

    // q A-fragments for K(d)=0..31 and 32..63, with 1/sqrt(DH)=0.125 folded in
    const h16* qbase = q + ((size_t)(b * NN + q0 + m) * CC_ + h * DH);
    v16h qa0 = mk16(*(const v8h*)(qbase + 16 * kg),
                    *(const v8h*)(qbase + 8 + 16 * kg));
    v16h qa1 = mk16(*(const v8h*)(qbase + 32 + 16 * kg),
                    *(const v8h*)(qbase + 40 + 16 * kg));
#pragma unroll
    for (int i = 0; i < 16; ++i) {        // exact: power-of-two scale in f16
        qa0[i] = qa0[i] * (h16)0.125f;
        qa1[i] = qa1[i] * (h16)0.125f;
    }

    float rm[8], rs[8];
#pragma unroll
    for (int j = 0; j < 8; ++j) { rm[j] = -__builtin_inff(); rs[j] = 0.0f; }
    v8f zacc0 = {}, zacc1 = {}, zacc2 = {}, zacc3 = {};

    for (int j0 = 0; j0 < NN; j0 += 32) {
        // ---- scores (pre-scaled): 16 x 32 (two 16x16 tiles), K = 64 ----
        v8f s0 = {}, s1 = {};
        {
            const h16* kb0 = k + ((size_t)(b * NN + j0 + m) * CC_ + h * DH + 16 * kg);
            const h16* kb1 = k + ((size_t)(b * NN + j0 + 16 + m) * CC_ + h * DH + 16 * kg);
            s0 = wmma16(qa0, *(const v16h*)(kb0), s0);
            s0 = wmma16(qa1, *(const v16h*)(kb0 + 32), s0);
            s1 = wmma16(qa0, *(const v16h*)(kb1), s1);
            s1 = wmma16(qa1, *(const v16h*)(kb1 + 32), s1);
        }
        // ---- online softmax (row m = j + 8*kg lives on a 16-lane group) ----
        float alpha[8];
#pragma unroll
        for (int j = 0; j < 8; ++j) {
            float a0 = s0[j];
            float a1 = s1[j];
            float tm = grp_max16(fmaxf(a0, a1));
            float nm = fmaxf(rm[j], tm);
            float al = __expf(rm[j] - nm);
            float p0 = __expf(a0 - nm);
            float p1 = __expf(a1 - nm);
            float ts = grp_sum16(p0 + p1);
            rs[j] = rs[j] * al + ts;
            rm[j] = nm;
            alpha[j] = al;
            // C-layout -> row-major LDS (A-layout friendly)
            myp[(j + 8 * kg) * 32 + m]      = (h16)p0;
            myp[(j + 8 * kg) * 32 + 16 + m] = (h16)p1;
        }
        asm volatile("s_wait_dscnt 0x0" ::: "memory");

        // A-fragment of P: rows m, K = 32 keys
    const v16h pa = mk16(*(const v8h*)(myp + m * 32 + 16 * kg),
                             *(const v8h*)(myp + m * 32 + 8 + 16 * kg));

        // rescale running accumulators
#pragma unroll
        for (int j = 0; j < 8; ++j) {
            zacc0[j] *= alpha[j]; zacc1[j] *= alpha[j];
            zacc2[j] *= alpha[j]; zacc3[j] *= alpha[j];
        }
        // ---- z += P(16x32) * V(32x64); V pre-transposed: vT[(b*768+c)*1024 + key] ----
        const size_t vrow = (size_t)(b * CC_ + h * DH + m) * NN + j0 + 16 * kg;
        zacc0 = wmma16(pa, *(const v16h*)(vT + vrow), zacc0);
        zacc1 = wmma16(pa, *(const v16h*)(vT + vrow + (size_t)16 * NN), zacc1);
        zacc2 = wmma16(pa, *(const v16h*)(vT + vrow + (size_t)32 * NN), zacc2);
        zacc3 = wmma16(pa, *(const v16h*)(vT + vrow + (size_t)48 * NN), zacc3);
    }

    // ---- epilogue: z /= rowsum, store row-major f16 [b,n,h*64+d] ----
    v8f zs[4] = {zacc0, zacc1, zacc2, zacc3};
#pragma unroll
    for (int s = 0; s < 4; ++s) {
#pragma unroll
        for (int j = 0; j < 8; ++j) {
            const int row = b * NN + q0 + j + 8 * kg;
            const int col = h * DH + s * 16 + m;
            z[(size_t)row * CC_ + col] = (h16)(zs[s][j] / rs[j]);
        }
    }
}

extern "C" void kernel_launch(void* const* d_in, const int* in_sizes, int n_in,
                              void* d_out, int out_size, void* d_ws, size_t ws_size,
                              hipStream_t stream) {
    const float* x   = (const float*)d_in[0];
    const float* Wq  = (const float*)d_in[1];
    const float* bq  = (const float*)d_in[2];
    const float* Wk  = (const float*)d_in[3];
    const float* bk  = (const float*)d_in[4];
    const float* Wv  = (const float*)d_in[5];
    const float* bv  = (const float*)d_in[6];
    const float* Wo  = (const float*)d_in[7];
    const float* bo  = (const float*)d_in[8];
    const float* g1  = (const float*)d_in[9];
    const float* b1  = (const float*)d_in[10];
    const float* g2  = (const float*)d_in[11];
    const float* b2  = (const float*)d_in[12];
    const float* W1  = (const float*)d_in[13];
    const float* bf1 = (const float*)d_in[14];
    const float* W2  = (const float*)d_in[15];
    const float* bf2 = (const float*)d_in[16];

    const size_t BNC = (size_t)BB * NN * CC_;       // 6,291,456
    const size_t WCC = (size_t)CC_ * CC_;           //   589,824
    const size_t WCF = (size_t)CC_ * FF_;           // 2,359,296

    h16* xn  = (h16*)d_ws;        // later reused as z
    h16* qb  = xn + BNC;          // later reused as bn
    h16* kbf = qb + BNC;          // later reused (with vT) as h
    h16* vT  = kbf + BNC;
    float* bbuf = (float*)(vT + BNC);
    h16* WqT = (h16*)(bbuf + BNC);
    h16* WkT = WqT + WCC;
    h16* WvT = WkT + WCC;
    h16* WoT = WvT + WCC;
    h16* W1T = WoT + WCC;
    h16* W2T = W1T + WCF;
    h16* zb = xn;
    h16* bn = qb;
    h16* hb = kbf;

    const dim3 blk(128);
    const dim3 g768(256, 3);    // R=8192 -> 256 M32-tiles; N=768 -> 12 n64-tiles / 4 waves
    const dim3 gff(256, 12);    // N=3072

    // weight transpose-casts (f32 -> f16, N-major)
    castT_kernel<<<(int)((WCC + 255) / 256), 256, 0, stream>>>(Wq, WqT, CC_, CC_);
    castT_kernel<<<(int)((WCC + 255) / 256), 256, 0, stream>>>(Wk, WkT, CC_, CC_);
    castT_kernel<<<(int)((WCC + 255) / 256), 256, 0, stream>>>(Wv, WvT, CC_, CC_);
    castT_kernel<<<(int)((WCC + 255) / 256), 256, 0, stream>>>(Wo, WoT, CC_, CC_);
    castT_kernel<<<(int)((WCF + 255) / 256), 256, 0, stream>>>(W1, W1T, CC_, FF_);
    castT_kernel<<<(int)((WCF + 255) / 256), 256, 0, stream>>>(W2, W2T, FF_, CC_);

    // LN1
    ln_kernel<<<BB * NN, 256, 0, stream>>>(x, g1, b1, xn);

    // QKV projections
    gemm_kernel<MODE_F16><<<g768, blk, 0, stream>>>(xn, WqT, bq, qb, nullptr, nullptr, CC_, CC_);
    gemm_kernel<MODE_F16><<<g768, blk, 0, stream>>>(xn, WkT, bk, kbf, nullptr, nullptr, CC_, CC_);
    gemm_kernel<MODE_TRANS_F16><<<g768, blk, 0, stream>>>(xn, WvT, bv, vT, nullptr, nullptr, CC_, CC_);

    // attention (writes zb == xn, which is free now)
    attn_kernel<<<1536, blk, 0, stream>>>(qb, kbf, vT, zb);

    // output projection + residual -> bbuf (f32)
    gemm_kernel<MODE_F32_RES><<<g768, blk, 0, stream>>>(zb, WoT, bo, nullptr, bbuf, x, CC_, CC_);

    // LN2 -> bn (reuses q buffer)
    ln_kernel<<<BB * NN, 256, 0, stream>>>(bbuf, g2, b2, bn);

    // FC1 + ReLU -> hb (reuses k+vT region)
    gemm_kernel<MODE_F16_RELU><<<gff, blk, 0, stream>>>(bn, W1T, bf1, hb, nullptr, nullptr, CC_, FF_);

    // FC2 + bias + residual -> d_out (f32)
    gemm_kernel<MODE_F32_RES><<<g768, blk, 0, stream>>>(hb, W2T, bf2, nullptr, (float*)d_out, bbuf, FF_, CC_);
}
